// LatentNCA_59751585022280
// MI455X (gfx1250) — matrix operand order repaired and checked by reading, not compiled, not run on hardware
//
#include <hip/hip_runtime.h>

typedef __attribute__((ext_vector_type(16))) _Float16 v16h;
typedef __attribute__((ext_vector_type(8)))  float    v8f;
typedef _Float16 half_t;

#define NB    4
#define CCH   16
#define IMGH  128
#define IMGW  128
#define HID   128
#define FIN_P 64      // 48 padded to 64
#define FEAT_STRIDE 72
#define H_STRIDE    132

// LDS carve (bytes)
#define XS_BYTES    (18*18*16*4)            // 20736 (also reused as 16KB out-staging)
#define FEAT_BYTES  (256*FEAT_STRIDE*2)     // 36864
#define HBUF_BYTES  (256*H_STRIDE*2)        // 67584
#define WPACK_HALFS (FIN_P*HID + HID*HID + HID*CCH)   // 26624
#define WPACK_BYTES (WPACK_HALFS*2)         // 53248
#define SMEM_BYTES  (XS_BYTES + FEAT_BYTES + HBUF_BYTES + WPACK_BYTES)  // 178432

// wpack regions (halfs), all stored in WMMA B-fragment order:
//   frag f: 512 halfs; lane ln (0..31) owns halfs [f*512 + ln*16, +16)
//   element e of lane ln = W[k = kt*32 + e + 16*(ln>>4)][col = n*16 + (ln&15)]
#define W1F_OFF  0        // 16 frags  (n=0..7, kt=0..1)  -> 8192 halfs
#define W2F_OFF  8192     // 32 frags  (n=0..7, kt=0..3)  -> 16384 halfs
#define W3F_OFF  24576    // 4 frags   (kt=0..3)          -> 2048 halfs

// workspace layout (bytes)
#define WS_FILM   0                         // 4 batches * 4 vecs * 128 f32 = 8192 B
#define WS_WPACK  8192                      // 53248 B
#define WS_XA     65536                     // 4 MB
#define WS_XB     (WS_XA + 4u*CCH*IMGH*IMGW*4u)

// ---------------------------------------------------------------------------
// Prep: FiLM params (blocks 0..3) + B-fragment-swizzled f16 weights (4..107)
// ---------------------------------------------------------------------------
__global__ __launch_bounds__(256) void nca_prep_kernel(
    const int* __restrict__ cond, const float* __restrict__ z,
    const float* __restrict__ embed_W,
    const float* __restrict__ f1W, const float* __restrict__ f1b,
    const float* __restrict__ f2W, const float* __restrict__ f2b,
    const float* __restrict__ fc1W, const float* __restrict__ fc2W,
    const float* __restrict__ fc3W,
    float* __restrict__ film, half_t* __restrict__ wpack)
{
    if (blockIdx.x < 4) {
        const int bb = blockIdx.x;
        const int o  = threadIdx.x;           // 0..255
        const int ci = cond[bb];
        float acc1 = f1b[o];
        float acc2 = f2b[o];
        #pragma unroll 4
        for (int j = 0; j < 80; ++j) {
            float cz = (j < 64) ? embed_W[ci*64 + j] : z[bb*16 + (j - 64)];
            acc1 += f1W[o*80 + j] * cz;
            acc2 += f2W[o*80 + j] * cz;
        }
        if (o < 128) {   // gamma halves
            film[(bb*4 + 0)*HID + o] = acc1;
            film[(bb*4 + 2)*HID + o] = acc2;
        } else {         // beta halves
            film[(bb*4 + 1)*HID + (o - 128)] = acc1;
            film[(bb*4 + 3)*HID + (o - 128)] = acc2;
        }
    } else {
        int idx = (blockIdx.x - 4)*256 + threadIdx.x;   // 0..26623
        float v;
        if (idx < W2F_OFF) {                            // W1 fragments
            int f  = idx >> 9;                          // 0..15
            int n  = f >> 1, kt = f & 1;
            int r  = idx & 511;
            int ln = r >> 4, e = r & 15;
            int k  = kt*32 + e + 16*(ln >> 4);
            int o  = n*16 + (ln & 15);
            v = (k < 48) ? fc1W[o*48 + k] : 0.0f;       // fc1_W is (128,48)
        } else if (idx < W3F_OFF) {                     // W2 fragments
            int t  = idx - W2F_OFF;
            int f  = t >> 9;                            // 0..31
            int n  = f >> 2, kt = f & 3;
            int r  = t & 511;
            int ln = r >> 4, e = r & 15;
            int k  = kt*32 + e + 16*(ln >> 4);
            int o  = n*16 + (ln & 15);
            v = fc2W[o*128 + k];                        // fc2_W is (128,128)
        } else {                                        // W3 fragments
            int t  = idx - W3F_OFF;
            int kt = t >> 9;                            // 0..3
            int r  = t & 511;
            int ln = r >> 4, e = r & 15;
            int k  = kt*32 + e + 16*(ln >> 4);
            int c  = ln & 15;
            v = fc3W[c*128 + k];                        // fc3_W is (16,128)
        }
        wpack[idx] = (half_t)v;
    }
}

// ---------------------------------------------------------------------------
// One NCA step: block = 16x16 spatial tile of one batch image.
// ---------------------------------------------------------------------------
__global__ __launch_bounds__(256) void nca_step_kernel(
    const float* __restrict__ xin, float* __restrict__ xout,
    const half_t* __restrict__ wpack_g, const float* __restrict__ film,
    const float* __restrict__ fc1b, const float* __restrict__ fc2b,
    const float* __restrict__ fc3b)
{
    extern __shared__ char smem_raw[];
    float*  xs   = (float*)(smem_raw);                                  // [18*18][16]
    half_t* feat = (half_t*)(smem_raw + XS_BYTES);                      // [256][72]
    half_t* hbuf = (half_t*)(smem_raw + XS_BYTES + FEAT_BYTES);         // [256][132]
    half_t* wsm  = (half_t*)(smem_raw + XS_BYTES + FEAT_BYTES + HBUF_BYTES);

    const int tid  = threadIdx.x;
    const int lane = tid & 31;
    const int wave = tid >> 5;
    const int bw   = blockIdx.x;     // tile col
    const int bh   = blockIdx.y;     // tile row
    const int b    = blockIdx.z;     // batch
    const int h0   = bh * 16, w0 = bw * 16;

    // --- cache f16 weight fragments in LDS (3328 x uint4 = 53248 B) ---
    {
        const uint4* src = (const uint4*)wpack_g;
        uint4*       dst = (uint4*)wsm;
        for (int i = tid; i < (int)(WPACK_BYTES/16); i += 256) dst[i] = src[i];
    }

    // --- load 18x18 halo of x, channel-major for coalescing (zero padded) ---
    for (int i = tid; i < 18*18*16; i += 256) {
        int c   = i / 324;
        int pos = i - c*324;
        int ly  = pos / 18, lx = pos - ly*18;
        int gh = h0 + ly - 1, gw = w0 + lx - 1;
        float v = 0.0f;
        if (gh >= 0 && gh < IMGH && gw >= 0 && gw < IMGW)
            v = xin[(((size_t)b*CCH + c)*IMGH + gh)*IMGW + gw];
        xs[pos*16 + c] = v;
    }
    __syncthreads();

    // --- feat = [x, sobel_x, sobel_y] as f16, K padded to 64 ---
    for (int i = tid; i < 256*16; i += 256) {
        int c = i & 15, p = i >> 4;
        int px = p & 15, py = p >> 4;
        int base = (py*18 + px)*16 + c;          // top-left of 3x3 window
        float x00 = xs[base +  0*16], x01 = xs[base +  1*16], x02 = xs[base +  2*16];
        float x10 = xs[base + 18*16], x11 = xs[base + 19*16], x12 = xs[base + 20*16];
        float x20 = xs[base + 36*16], x21 = xs[base + 37*16], x22 = xs[base + 38*16];
        float gx = (x02 - x00) + 2.0f*(x12 - x10) + (x22 - x20);
        float gy = (x20 - x00) + 2.0f*(x21 - x01) + (x22 - x02);
        half_t* f = feat + p*FEAT_STRIDE;
        f[c]      = (half_t)x11;
        f[16 + c] = (half_t)gx;
        f[32 + c] = (half_t)gy;
        f[48 + c] = (half_t)0.0f;                // K padding
    }
    __syncthreads();

    const v16h* w1f = (const v16h*)(wsm + W1F_OFF);   // [16 frags][32 lanes]
    const v16h* w2f = (const v16h*)(wsm + W2F_OFF);   // [32 frags][32 lanes]
    const v16h* w3f = (const v16h*)(wsm + W3F_OFF);   // [ 4 frags][32 lanes]

    const int m  = lane & 15;                    // row (A) / col (B,D) index
    const int hi = lane >> 4;

    float outv[16];                              // staged results across barrier

    #pragma unroll
    for (int i = 0; i < 2; ++i) {
        const int mt  = wave*2 + i;              // M tile (16 pixels)
        const int pix = mt*16 + m;

        // ---- layer 1: feat(16x48) @ W1t -> h1(16x128) ----
        v16h a0, a1;
        #pragma unroll
        for (int e = 0; e < 16; ++e) {
            int k = e + 16*hi;                   // contiguous k order (matches B frags)
            a0[e] = feat[pix*FEAT_STRIDE + k];
            a1[e] = feat[pix*FEAT_STRIDE + 32 + k];
        }
        #pragma unroll
        for (int n = 0; n < 8; ++n) {
            v16h b0 = w1f[(n*2 + 0)*32 + lane];
            v16h b1 = w1f[(n*2 + 1)*32 + lane];
            v8f acc = {};
            acc = __builtin_amdgcn_wmma_f32_16x16x32_f16(false, a0, false, b0, (short)0, acc, false, false);
            acc = __builtin_amdgcn_wmma_f32_16x16x32_f16(false, a1, false, b1, (short)0, acc, false, false);
            const int o = n*16 + m;
            const float bias = fc1b[o];
            const float g  = film[(b*4 + 0)*HID + o];
            const float bt = film[(b*4 + 1)*HID + o];
            #pragma unroll
            for (int v = 0; v < 8; ++v) {
                float hv = acc[v] + bias;
                hv = hv > 0.0f ? hv : 0.0f;
                hv = g*hv + bt;
                hbuf[(mt*16 + v + 8*hi)*H_STRIDE + o] = (half_t)hv;
            }
        }

        // ---- layer 2: h1(16x128) @ W2t -> h2(16x128), in-place in hbuf ----
        v16h a2[4];
        #pragma unroll
        for (int kt = 0; kt < 4; ++kt)
            #pragma unroll
            for (int e = 0; e < 16; ++e) {
                int k = kt*32 + e + 16*hi;
                a2[kt][e] = hbuf[pix*H_STRIDE + k];
            }
        #pragma unroll
        for (int n = 0; n < 8; ++n) {
            v8f acc = {};
            #pragma unroll
            for (int kt = 0; kt < 4; ++kt) {
                v16h bf = w2f[(n*4 + kt)*32 + lane];
                acc = __builtin_amdgcn_wmma_f32_16x16x32_f16(false, a2[kt], false, bf, (short)0, acc, false, false);
            }
            const int o = n*16 + m;
            const float bias = fc2b[o];
            const float g  = film[(b*4 + 2)*HID + o];
            const float bt = film[(b*4 + 3)*HID + o];
            #pragma unroll
            for (int v = 0; v < 8; ++v) {
                float hv = acc[v] + bias;
                hv = hv > 0.0f ? hv : 0.0f;
                hv = g*hv + bt;
                hbuf[(mt*16 + v + 8*hi)*H_STRIDE + o] = (half_t)hv;
            }
        }

        // ---- layer 3: h2(16x128) @ W3t -> dx(16x16), Euler update ----
        v16h a3[4];
        #pragma unroll
        for (int kt = 0; kt < 4; ++kt)
            #pragma unroll
            for (int e = 0; e < 16; ++e) {
                int k = kt*32 + e + 16*hi;
                a3[kt][e] = hbuf[pix*H_STRIDE + k];
            }
        v8f acc = {};
        #pragma unroll
        for (int kt = 0; kt < 4; ++kt) {
            v16h bf = w3f[kt*32 + lane];
            acc = __builtin_amdgcn_wmma_f32_16x16x32_f16(false, a3[kt], false, bf, (short)0, acc, false, false);
        }
        const float bias3 = fc3b[m];             // lane's output channel
        #pragma unroll
        for (int v = 0; v < 8; ++v) {
            int p  = mt*16 + v + 8*hi;
            int py = p >> 4, px = p & 15;
            float dx = acc[v] + bias3;
            dx = fminf(fmaxf(dx, -10.0f), 10.0f);
            float xold = xs[((py + 1)*18 + (px + 1))*16 + m];
            outv[i*8 + v] = xold + 0.1f*dx;
        }
    }

    // --- transpose results through LDS (reuse xs) for coalesced NCHW store ---
    __syncthreads();                             // all xold reads done
    float* ob = xs;                              // [16 ch][256 px], 16 KB
    #pragma unroll
    for (int i = 0; i < 2; ++i) {
        const int mt = wave*2 + i;
        #pragma unroll
        for (int v = 0; v < 8; ++v)
            ob[m*256 + mt*16 + v + 8*hi] = outv[i*8 + v];
    }
    __syncthreads();
    for (int i = tid; i < CCH*256; i += 256) {
        int c = i >> 8, p = i & 255;
        int py = p >> 4, px = p & 15;
        xout[(((size_t)b*CCH + c)*IMGH + (h0 + py))*IMGW + (w0 + px)] = ob[c*256 + p];
    }
}

// ---------------------------------------------------------------------------
extern "C" void kernel_launch(void* const* d_in, const int* in_sizes, int n_in,
                              void* d_out, int out_size, void* d_ws, size_t ws_size,
                              hipStream_t stream) {
    const float* x       = (const float*)d_in[0];
    const int*   cond    = (const int*)  d_in[1];
    const float* z       = (const float*)d_in[2];
    const float* embed_W = (const float*)d_in[3];
    const float* f1W     = (const float*)d_in[4];
    const float* f1b     = (const float*)d_in[5];
    const float* f2W     = (const float*)d_in[6];
    const float* f2b     = (const float*)d_in[7];
    const float* fc1W    = (const float*)d_in[8];
    const float* fc1b    = (const float*)d_in[9];
    const float* fc2W    = (const float*)d_in[10];
    const float* fc2b    = (const float*)d_in[11];
    const float* fc3W    = (const float*)d_in[12];
    const float* fc3b    = (const float*)d_in[13];
    (void)in_sizes; (void)n_in; (void)out_size; (void)ws_size;

    char* ws = (char*)d_ws;
    float*  film  = (float*) (ws + WS_FILM);
    half_t* wpack = (half_t*)(ws + WS_WPACK);
    float*  bufA  = (float*) (ws + WS_XA);
    float*  bufB  = (float*) (ws + WS_XB);
    float*  out   = (float*) d_out;

    nca_prep_kernel<<<dim3(108), 256, 0, stream>>>(
        cond, z, embed_W, f1W, f1b, f2W, f2b, fc1W, fc2W, fc3W, film, wpack);

    const int n_steps = 12;                       // reference: fixed scalar
    dim3 grid(IMGW/16, IMGH/16, NB);
    for (int s = 0; s < n_steps; ++s) {
        const float* in  = (s == 0) ? x : ((s & 1) ? bufA : bufB);
        float*       o   = (s == n_steps - 1) ? out : ((s & 1) ? bufB : bufA);
        nca_step_kernel<<<grid, 256, SMEM_BYTES, stream>>>(
            in, o, wpack, film, fc1b, fc2b, fc3b);
    }
}
